// _GraphLearning_21217138442723
// MI455X (gfx1250) — matrix operand order repaired and checked by reading, not compiled, not run on hardware
//
#include <hip/hip_runtime.h>
#include <hip/hip_bf16.h>
#include <math.h>

#define NN 8192
#define DD 64
#define TOPK 32
#define ALPHA_C 3.0f

typedef __attribute__((ext_vector_type(16))) _Float16 v16h;
typedef __attribute__((ext_vector_type(8)))  _Float16 v8h;
typedef __attribute__((ext_vector_type(8)))  float    v8f;

// Branch-free tanh: 1 - 2/(exp(2x)+1). Native v_exp_f32 saturates to +inf/0
// at the extremes, so t -> +/-1 exactly where it matters; no EXEC divergence.
__device__ inline float fast_tanh(float x) {
  float e = __expf(2.0f * x);                      // v_mul + v_exp_f32
  return 1.0f - 2.0f * __builtin_amdgcn_rcpf(e + 1.0f);  // v_rcp_f32
}

// ---------------------------------------------------------------------------
// Kernel A: h = tanh(ALPHA * (emb[idx] @ W^T + b)) -> f16, for both branches.
// Block = 256 threads handles 4 rows; W staged transposed in LDS so the inner
// product reads Wt[k*D+o] (bank = o, conflict-free across the row group).
// ---------------------------------------------------------------------------
__global__ __launch_bounds__(256) void mlp_kernel(
    const int* __restrict__ idx,
    const float* __restrict__ emb1, const float* __restrict__ emb2,
    const float* __restrict__ W1, const float* __restrict__ b1,
    const float* __restrict__ W2, const float* __restrict__ b2,
    _Float16* __restrict__ h1, _Float16* __restrict__ h2) {
  __shared__ float Wt1[DD * DD];   // Wt1[k*D + o] = W1[o*D + k]
  __shared__ float Wt2[DD * DD];
  __shared__ float xs[4][DD];
  __shared__ float ys[4][DD];

  const int tid = threadIdx.x;
  for (int i = tid; i < DD * DD; i += 256) {
    int o = i >> 6, k = i & 63;
    Wt1[k * DD + o] = W1[i];
    Wt2[k * DD + o] = W2[i];
  }

  const int sub = tid >> 6;       // 0..3 : row within block
  const int o   = tid & 63;       // output feature
  const int row = blockIdx.x * 4 + sub;
  const int g   = idx[row];
  xs[sub][o] = emb1[(size_t)g * DD + o];
  ys[sub][o] = emb2[(size_t)g * DD + o];
  __syncthreads();

  float s1 = b1[o];
  float s2 = b2[o];
#pragma unroll 8
  for (int k = 0; k < DD; ++k) {
    s1 += xs[sub][k] * Wt1[k * DD + o];
    s2 += ys[sub][k] * Wt2[k * DD + o];
  }
  h1[(size_t)row * DD + o] = (_Float16)fast_tanh(ALPHA_C * s1);
  h2[(size_t)row * DD + o] = (_Float16)fast_tanh(ALPHA_C * s2);
}

// ---------------------------------------------------------------------------
// Kernel B: A = relu(tanh(ALPHA * (h1 @ h2^T - h2 @ h1^T))) via WMMA f16->f32.
// One wave per 16x16 output tile, 8 waves per block span a 16x128 strip.
// Fragments use contiguous k = 16*(lane/16) + s for A and B (16-bit A 16x32, B 32x16).
// ---------------------------------------------------------------------------
__device__ inline v16h load_fragA(const _Float16* __restrict__ h, int m0,
                                  int lane, int ks) {
  const int m  = m0 + (lane & 15);
  const int cb = ((lane >> 4) << 4) + ks;          // K base: 0 or 16 (+ks)
  const _Float16* p = h + (size_t)m * DD + cb;
  v8h lo = *(const v8h*)(p);                       // K = cb .. cb+7
  v8h hi = *(const v8h*)(p + 8);                   // K = cb+8 .. cb+15
  return __builtin_shufflevector(lo, hi, 0, 1, 2, 3, 4, 5, 6, 7,
                                 8, 9, 10, 11, 12, 13, 14, 15);
}

__device__ inline v16h load_fragB(const _Float16* __restrict__ h, int n0,
                                  int lane, int ks) {
  const int j  = n0 + (lane & 15);                 // output column
  const int kb = ((lane >> 4) << 4) + ks;          // K base: 0 or 16 (+ks)
  return *(const v16h*)(h + (size_t)j * DD + kb);  // 32B-aligned contiguous
}

__global__ __launch_bounds__(256) void pairwise_wmma_kernel(
    const _Float16* __restrict__ h1, const _Float16* __restrict__ h2,
    float* __restrict__ out) {
  const int lane = threadIdx.x & 31;
  const int wave = threadIdx.x >> 5;
  const int m0 = blockIdx.y * 16;
  const int n0 = blockIdx.x * 128 + wave * 16;

  v8f acc1 = {};   // v1 . v2^T
  v8f acc2 = {};   // v2 . v1^T
#pragma unroll
  for (int ks = 0; ks < DD; ks += 32) {
    v16h a1 = load_fragA(h1, m0, lane, ks);
    v16h a2 = load_fragA(h2, m0, lane, ks);
    v16h b1 = load_fragB(h1, n0, lane, ks);
    v16h b2 = load_fragB(h2, n0, lane, ks);
    acc1 = __builtin_amdgcn_wmma_f32_16x16x32_f16(false, a1, false, b2,
                                                  (short)0, acc1, false, false);
    acc2 = __builtin_amdgcn_wmma_f32_16x16x32_f16(false, a2, false, b1,
                                                  (short)0, acc2, false, false);
  }

  // C/D layout: VGPR r, lane -> M = m0 + r + (lane>>4)*8, N = n0 + (lane&15)
  const int col   = n0 + (lane & 15);
  const int rbase = m0 + ((lane >> 4) << 3);
#pragma unroll
  for (int r = 0; r < 8; ++r) {
    float a = acc1[r] - acc2[r];
    float t = fast_tanh(ALPHA_C * a);
    float v = t > 0.0f ? t : 0.0f;
    out[(size_t)(rbase + r) * NN + col] = v;
  }
}

// ---------------------------------------------------------------------------
// Kernel C: per-row top-K mask. One block (256 threads) per row.
// Two-level selection: each thread owns the strided chunk {tid + 256*i},
// keeping its chunk-max in cmax[]. Each of the 32 rounds reduces the 256
// chunk maxima with wave32 shuffles (2 barriers), then only the winning
// thread rescans its 32 elements (tracking max + runner-up) to mark the
// selected entry and refresh its chunk max in one pass.
// ---------------------------------------------------------------------------
__global__ __launch_bounds__(256) void topk_mask_kernel(
    const float* __restrict__ noise, float* __restrict__ out) {
  __shared__ float score[NN];     // 32 KB
  __shared__ float cmax[256];
  __shared__ float wred_v[8];
  __shared__ int   wred_i[8];
  __shared__ float bcast_v;
  __shared__ int   bcast_t;

  const int row  = blockIdx.x;
  const int tid  = threadIdx.x;
  const int lane = tid & 31;
  const int wave = tid >> 5;
  const float MARK = -1e30f;
  const float* __restrict__ arow = out + (size_t)row * NN;
  const float* __restrict__ nrow = noise + (size_t)row * NN;

  // Build score row + per-thread chunk max (strided -> bank-conflict-free).
  float m = -1e29f;
#pragma unroll
  for (int i = 0; i < NN / 256; ++i) {
    int j = tid + 256 * i;
    float s = arow[j] + 0.01f * nrow[j];
    score[j] = s;
    if (s > m) m = s;
  }
  cmax[tid] = m;
  __syncthreads();

  float T = 0.0f;
  for (int it = 0; it < TOPK; ++it) {
    // Level 1: argmax over the 256 chunk maxima.
    float v = cmax[tid];
    int   ci = tid;
#pragma unroll
    for (int off = 16; off > 0; off >>= 1) {
      float ov = __shfl_xor(v, off, 32);
      int   oi = __shfl_xor(ci, off, 32);
      if (ov > v) { v = ov; ci = oi; }
    }
    if (lane == 0) { wred_v[wave] = v; wred_i[wave] = ci; }
    __syncthreads();
    if (wave == 0) {
      float v2 = (lane < 8) ? wred_v[lane] : -1e30f;
      int   c2 = (lane < 8) ? wred_i[lane] : 0;
#pragma unroll
      for (int off = 4; off > 0; off >>= 1) {
        float ov = __shfl_xor(v2, off, 32);
        int   oi = __shfl_xor(c2, off, 32);
        if (ov > v2) { v2 = ov; c2 = oi; }
      }
      if (lane == 0) { bcast_v = v2; bcast_t = c2; }
    }
    __syncthreads();
    T = bcast_v;                   // value of it-th largest; final = threshold
    // Level 2: winning thread marks its max and refreshes chunk max.
    if (tid == bcast_t) {
      float m1 = -1e29f, m2 = -1e29f;
      int bj = tid;
#pragma unroll
      for (int i = 0; i < NN / 256; ++i) {
        int j = tid + 256 * i;
        float s = score[j];
        if (s > m1) { m2 = m1; m1 = s; bj = j; }
        else if (s > m2) { m2 = s; }
      }
      score[bj] = MARK;
      cmax[tid] = m2;
    }
    __syncthreads();
  }

  // Emit masked row: marked entries + threshold ties keep their A value.
#pragma unroll
  for (int i = 0; i < NN / 256; ++i) {
    int j = tid + 256 * i;
    float s = score[j];
    bool sel = (s == MARK) || (s >= T);
    out[(size_t)row * NN + j] = sel ? arow[j] : 0.0f;
  }
}

// ---------------------------------------------------------------------------
extern "C" void kernel_launch(void* const* d_in, const int* in_sizes, int n_in,
                              void* d_out, int out_size, void* d_ws, size_t ws_size,
                              hipStream_t stream) {
  const int*   idx   = (const int*)d_in[0];
  const float* emb1  = (const float*)d_in[1];
  const float* emb2  = (const float*)d_in[2];
  const float* W1    = (const float*)d_in[3];
  const float* b1    = (const float*)d_in[4];
  const float* W2    = (const float*)d_in[5];
  const float* b2    = (const float*)d_in[6];
  const float* noise = (const float*)d_in[7];
  float* out = (float*)d_out;

  _Float16* h1 = (_Float16*)d_ws;                 // N*D f16 = 1 MB
  _Float16* h2 = h1 + (size_t)NN * DD;            // next 1 MB

  // Stage 1: gathered MLP + tanh saturation -> f16 activations (L2 resident)
  mlp_kernel<<<NN / 4, 256, 0, stream>>>(idx, emb1, emb2, W1, b1, W2, b2, h1, h2);

  // Stage 2: antisymmetric pairwise scores via WMMA, write full A (256 MB)
  dim3 gridB(NN / 128, NN / 16);                  // (64, 512), 8 waves/block
  pairwise_wmma_kernel<<<gridB, 256, 0, stream>>>(h1, h2, out);

  // Stage 3: per-row top-K threshold + mask (in-place on d_out)
  topk_mask_kernel<<<NN, 256, 0, stream>>>(noise, out);
}
